// QuantumModulatedEncoder_15101105013467
// MI455X (gfx1250) — matrix operand order repaired and checked by reading, not compiled, not run on hardware
//
#include <hip/hip_runtime.h>
#include <hip/hip_bf16.h>

// ---------------------------------------------------------------------------
// Problem constants (from reference)
// ---------------------------------------------------------------------------
#define IN_F 700
#define REDC 256
#define WID  4096
#define CLS  20
#define TT   100
#define BB   64
#define MM   (BB * TT)      // 6400
#define KP_SPATIAL 704      // 700 padded to multiple of 32

typedef __attribute__((ext_vector_type(16))) _Float16 v16h;
typedef __attribute__((ext_vector_type(8)))  _Float16 v8h;
typedef __attribute__((ext_vector_type(8)))  float    v8f;

union U16 { v16h v; v8h h[2]; };

// ---------------------------------------------------------------------------
// Preprocessing kernels
// ---------------------------------------------------------------------------
__global__ void absmax_kernel(const float* __restrict__ src, int n, unsigned int* out) {
    __shared__ float red[256];
    float m = 0.f;
    for (int i = blockIdx.x * blockDim.x + threadIdx.x; i < n; i += gridDim.x * blockDim.x)
        m = fmaxf(m, fabsf(src[i]));
    red[threadIdx.x] = m;
    __syncthreads();
    for (int s = 128; s > 0; s >>= 1) {
        if ((int)threadIdx.x < s) red[threadIdx.x] = fmaxf(red[threadIdx.x], red[threadIdx.x + s]);
        __syncthreads();
    }
    if (threadIdx.x == 0) atomicMax(out, __float_as_uint(red[0]));  // valid: all >= 0
}

// INT4 STE quantize (forward value) + cast to f16, zero-padding K -> Kp
__global__ void quant_pad_kernel(const float* __restrict__ W, _Float16* __restrict__ dst,
                                 const unsigned int* __restrict__ amax, int N, int K, int Kp) {
    int idx = blockIdx.x * 256 + threadIdx.x;
    if (idx >= N * Kp) return;
    int n = idx / Kp, k = idx - n * Kp;
    float scale = fmaxf(__uint_as_float(*amax) / 7.f, 1e-8f);
    float v = 0.f;
    if (k < K) {
        float w = W[(size_t)n * K + k];
        v = rintf(w / scale) * scale;   // round-half-even like jnp.round
    }
    dst[idx] = (_Float16)v;
}

// fp32 activations -> f16, zero-padding K -> Kp (for x)
__global__ void cvt_pad_f16_kernel(const float* __restrict__ src, _Float16* __restrict__ dst,
                                   int M, int K, int Kp) {
    int idx = blockIdx.x * 256 + threadIdx.x;
    if (idx >= M * Kp) return;
    int m = idx / Kp, k = idx - m * Kp;
    dst[idx] = (k < K) ? (_Float16)src[(size_t)m * K + k] : (_Float16)0.f;
}

// conv_W (O, I, 5) -> f16 B matrix [O][tap*256 + i]   (conv_W is NOT quantized in ref)
__global__ void pack_conv_kernel(const float* __restrict__ W, _Float16* __restrict__ dst) {
    int idx = blockIdx.x * 256 + threadIdx.x;
    if (idx >= REDC * REDC * 5) return;
    int o = idx / (REDC * 5), r = idx - o * (REDC * 5);
    int tap = r >> 8, i = r & 255;
    dst[idx] = (_Float16)W[(size_t)o * (REDC * 5) + i * 5 + tap];
}

// ---------------------------------------------------------------------------
// Tiled WMMA GEMM:  D = epilogue(A(MxK f16) * B(NxK f16, row-major over n)ᵀ + bias)
// Block = 256 threads (8 waves). Tile: M=128 (16 rows/wave), N=64 (4 frags), K-step 32.
// Double-buffered GLOBAL_LOAD_ASYNC_TO_LDS_B128 pipeline: exactly 3 async ops per
// wave per tile, so s_wait_asynccnt 0x3 retires tile i while tile i+1 is in flight.
// ---------------------------------------------------------------------------
enum { AMODE_DIRECT = 0, AMODE_CONV = 1 };
enum { EPI_RELU_F16 = 0, EPI_DRIVE = 1 };

__device__ __forceinline__ float hash_u01(unsigned int x) {
    x ^= x >> 16; x *= 0x7feb352du; x ^= x >> 15; x *= 0x846ca68bu; x ^= x >> 16;
    return (float)(x >> 8) * (1.0f / 16777216.0f);
}

// Flat shared-aperture address -> 32-bit LDS byte offset (low 32 bits of the address)
__device__ __forceinline__ unsigned int lds_off(const void* p) {
    return (unsigned int)(unsigned long long)p;
}

// 32B memory->LDS async copy (two b128; inst offset applies to both addresses)
__device__ __forceinline__ void async_copy_32B(unsigned int lds, const _Float16* gp) {
    asm volatile("global_load_async_to_lds_b128 %0, %1, off\n\t"
                 "global_load_async_to_lds_b128 %0, %1, off offset:16"
                 :: "v"(lds), "v"(gp) : "memory");
}
__device__ __forceinline__ void async_copy_16B(unsigned int lds, const _Float16* gp) {
    asm volatile("global_load_async_to_lds_b128 %0, %1, off"
                 :: "v"(lds), "v"(gp) : "memory");
}
__device__ __forceinline__ void wait_async3() {   // oldest tile landed, next in flight
    asm volatile("s_wait_asynccnt 0x3" ::: "memory");
}
__device__ __forceinline__ void wait_async0() {
    asm volatile("s_wait_asynccnt 0x0" ::: "memory");
}

template <int AMODE, int EPI>
__global__ __launch_bounds__(256)
void wmma_gemm_kernel(const _Float16* __restrict__ A, const _Float16* __restrict__ B,
                      const float* __restrict__ bias,
                      const _Float16* __restrict__ zguard,   // 192B of zeros (conv pad)
                      _Float16* __restrict__ Ch,             // f16 output (EPI_RELU_F16)
                      float* __restrict__ drive, float* __restrict__ spikes,  // EPI_DRIVE
                      const float* __restrict__ latent, const float* __restrict__ gain,
                      int M, int N, int K) {
    __shared__ __align__(16) _Float16 lA[2][128][40];
    __shared__ __align__(16) _Float16 lB[2][64][40];

    const int tid  = threadIdx.x;
    const int lane = tid & 31;
    const int wave = tid >> 5;
    const int mBase = blockIdx.y * 128;
    const int nBase = blockIdx.x * 64;

    v8f acc[4] = {};

    const int aRow = tid >> 1;            // 0..127
    const int aC0  = (tid & 1) * 16;      // 0 or 16
    const int bRow = tid >> 2;            // 0..63
    const int bC0  = (tid & 3) * 8;       // 0,8,16,24

    const unsigned int ldsA[2] = { lds_off(&lA[0][aRow][aC0]), lds_off(&lA[1][aRow][aC0]) };
    const unsigned int ldsB[2] = { lds_off(&lB[0][bRow][bC0]), lds_off(&lB[1][bRow][bC0]) };

    // Precompute conv-gather row decomposition (uniform per thread).
    int cvBase = 0, cvT = 0;
    if (AMODE == AMODE_CONV) {
        const int gm = mBase + aRow;
        const int bI = gm / TT;
        cvT = gm - bI * TT;
        cvBase = bI * TT;
    }

    // Stage one K-tile into LDS buffer `buf` (always exactly 3 async ops).
    auto stage = [&](int k0, int buf) {
        const _Float16* gpA;
        if (AMODE == AMODE_DIRECT) {
            gpA = A + (size_t)(mBase + aRow) * K + k0 + aC0;
        } else {
            // k = tap*256 + ci ; the 16-wide chunk never straddles a tap (256 % 32 == 0)
            const int tap = (k0 + aC0) >> 8;
            const int ci0 = (k0 + aC0) & 255;
            const int ts  = cvT + tap - 2;
            const bool ok = (ts >= 0) && (ts < TT);
            gpA = ok ? (A + ((size_t)(cvBase + ts)) * REDC + ci0) : zguard;  // branchless pad
        }
        async_copy_32B(ldsA[buf], gpA);
        async_copy_16B(ldsB[buf], B + (size_t)(nBase + bRow) * K + k0 + bC0);
    };

    const int nk = K >> 5;
    stage(0, 0);                               // prologue: tile 0 in flight (cnt=3)

    for (int i = 0; i < nk; ++i) {
        const int p = i & 1;
        if (i + 1 < nk) {
            stage((i + 1) << 5, p ^ 1);        // cnt = 6
            wait_async3();                     // tile i landed; tile i+1 in flight
        } else {
            wait_async0();                     // last tile
        }
        __syncthreads();

        // ---- fragments + 4 WMMAs per wave ----
        const int l15  = lane & 15;
        const int half = lane >> 4;
        U16 afrag;
        {
            // A 16x32 f16 layout: lane holds K chunks [8h, 8h+8) and [16+8h, 16+8h+8)
            const _Float16* ap = &lA[p][wave * 16 + l15][0];
            afrag.h[0] = *reinterpret_cast<const v8h*>(ap + half * 8);
            afrag.h[1] = *reinterpret_cast<const v8h*>(ap + 16 + half * 8);
        }
#pragma unroll
        for (int f = 0; f < 4; ++f) {
            // B 32x16 f16 layout: lane = column n, K chunks [8h, 8h+8) and [16+8h, 16+8h+8)
            U16 bfrag;
            const _Float16* bp = &lB[p][f * 16 + l15][0];
            bfrag.h[0] = *reinterpret_cast<const v8h*>(bp + half * 8);
            bfrag.h[1] = *reinterpret_cast<const v8h*>(bp + 16 + half * 8);
            acc[f] = __builtin_amdgcn_wmma_f32_16x16x32_f16(
                false, afrag.v, false, bfrag.v, (short)0, acc[f], false, false);
        }
        __syncthreads();   // buffer p free for re-staging next iteration
    }

    // ---- epilogue: C/D layout -> lane = (mHalf, n), VGPR v -> m = 8*mHalf + v ----
    const int nIdx  = lane & 15;
    const int mHalf = lane >> 4;
#pragma unroll
    for (int f = 0; f < 4; ++f) {
        const int n  = nBase + f * 16 + nIdx;
        const float bv = bias[n];
#pragma unroll
        for (int v = 0; v < 8; ++v) {
            const int m = mBase + wave * 16 + mHalf * 8 + v;   // always < M (exact tiling)
            float val = acc[f][v] + bv;
            const size_t off = (size_t)m * N + n;
            if (EPI == EPI_RELU_F16) {
                Ch[off] = (_Float16)fmaxf(val, 0.f);
            } else {
                float sp = (val > 20.f) ? val : log1pf(__expf(val));   // softplus
                int   t  = m % TT;
                float dv = sp * latent[t] * fabsf(gain[0]);
                drive[off] = dv;
                // spikes = min(Poisson(drive),1) == Bernoulli(1 - exp(-drive))
                float u = hash_u01((unsigned int)off * 2654435761u + 0x9E3779B9u);
                spikes[off] = (u >= __expf(-dv)) ? 1.0f : 0.0f;
            }
        }
    }
}

// ---------------------------------------------------------------------------
// weighted[b][w] = sum_t spikes[b,t,w] * exp(-2 + 2t/99)
// ---------------------------------------------------------------------------
__global__ void reduce_spikes_kernel(const float* __restrict__ spikes, float* __restrict__ weighted) {
    int idx = blockIdx.x * 256 + threadIdx.x;   // b*WID + w
    if (idx >= BB * WID) return;
    int b = idx / WID, w = idx - b * WID;
    float acc = 0.f;
#pragma unroll 4
    for (int t = 0; t < TT; ++t)
        acc += spikes[((size_t)(b * TT + t)) * WID + w] * __expf(-2.f + 2.f * t * (1.f / 99.f));
    weighted[idx] = acc;
}

// logits[b][c] = weighted[b] . q4(fc2_W[c]) + fc2_b[c]
__global__ void logits_kernel(const float* __restrict__ weighted, const float* __restrict__ fc2W,
                              const float* __restrict__ fc2b,
                              const unsigned int* __restrict__ amax, float* __restrict__ logits) {
    int b = blockIdx.x / CLS, c = blockIdx.x - b * CLS;
    float scale = fmaxf(__uint_as_float(amax[2]) / 7.f, 1e-8f);
    __shared__ float red[256];
    float acc = 0.f;
    for (int w = threadIdx.x; w < WID; w += 256) {
        float wq = rintf(fc2W[(size_t)c * WID + w] / scale) * scale;
        acc += weighted[(size_t)b * WID + w] * wq;
    }
    red[threadIdx.x] = acc;
    __syncthreads();
    for (int s = 128; s > 0; s >>= 1) {
        if ((int)threadIdx.x < s) red[threadIdx.x] += red[threadIdx.x + s];
        __syncthreads();
    }
    if (threadIdx.x == 0) logits[b * CLS + c] = red[0] + fc2b[c];
}

// ---------------------------------------------------------------------------
// Launcher
// ---------------------------------------------------------------------------
extern "C" void kernel_launch(void* const* d_in, const int* in_sizes, int n_in,
                              void* d_out, int out_size, void* d_ws, size_t ws_size,
                              hipStream_t stream) {
    const float* x         = (const float*)d_in[0];   // (B,T,IN_F)
    const float* latent    = (const float*)d_in[1];   // (T,)
    const float* spatial_W = (const float*)d_in[2];   // (RED, IN_F)
    const float* spatial_b = (const float*)d_in[3];
    const float* conv_W    = (const float*)d_in[4];   // (RED, RED, 5)
    const float* conv_b    = (const float*)d_in[5];
    const float* fc1_W     = (const float*)d_in[6];   // (WID, RED)
    const float* fc1_b     = (const float*)d_in[7];
    const float* fc2_W     = (const float*)d_in[8];   // (CLS, WID)
    const float* fc2_b     = (const float*)d_in[9];
    const float* gain      = (const float*)d_in[10];

    // Workspace carve-up (all offsets 256B aligned)
    char* ws = (char*)d_ws;
    unsigned int* absmax = (unsigned int*)ws;                    // [0, 64):  3 scales
    _Float16* zguard = (_Float16*)(ws + 64);                     // [64,256): zeros (conv pad)
    _Float16* WqS   = (_Float16*)(ws + 256);                     // 256*704*2  = 360448
    _Float16* WqF1  = (_Float16*)(ws + 360704);                  // 4096*256*2 = 2097152
    _Float16* convP = (_Float16*)(ws + 2457856);                 // 256*1280*2 = 655360
    _Float16* xh    = (_Float16*)(ws + 3113216);                 // 6400*704*2 = 9011200
    _Float16* xp    = (_Float16*)(ws + 12124416);                // 6400*256*2 = 3276800
    _Float16* cvo   = (_Float16*)(ws + 15401216);                // 6400*256*2 = 3276800
    float* weighted = (float*)(ws + 18678016);                   // 64*4096*4  = 1048576

    // Output layout: logits | spikes | drive
    float* logits     = (float*)d_out;
    float* spikes_out = logits + BB * CLS;
    float* drive_out  = spikes_out + (size_t)MM * WID;

    // --- zero absmax slots + zero-guard region in one memset ---
    hipMemsetAsync(ws, 0, 256, stream);

    // --- per-tensor abs-max for q4 scales ---
    absmax_kernel<<<64, 256, 0, stream>>>(spatial_W, REDC * IN_F, absmax + 0);
    absmax_kernel<<<256, 256, 0, stream>>>(fc1_W, WID * REDC, absmax + 1);
    absmax_kernel<<<64, 256, 0, stream>>>(fc2_W, CLS * WID, absmax + 2);

    // --- quantize / pack weights, convert x ---
    quant_pad_kernel<<<(REDC * KP_SPATIAL + 255) / 256, 256, 0, stream>>>(
        spatial_W, WqS, absmax + 0, REDC, IN_F, KP_SPATIAL);
    quant_pad_kernel<<<(WID * REDC + 255) / 256, 256, 0, stream>>>(
        fc1_W, WqF1, absmax + 1, WID, REDC, REDC);
    pack_conv_kernel<<<(REDC * REDC * 5 + 255) / 256, 256, 0, stream>>>(conv_W, convP);
    cvt_pad_f16_kernel<<<(MM * KP_SPATIAL + 255) / 256, 256, 0, stream>>>(
        x, xh, MM, IN_F, KP_SPATIAL);

    // --- GEMM 1: xp = relu(x @ q4(spatial_W)^T + b)   M=6400 K=704(pad) N=256 ---
    wmma_gemm_kernel<AMODE_DIRECT, EPI_RELU_F16><<<dim3(REDC / 64, MM / 128), 256, 0, stream>>>(
        xh, WqS, spatial_b, zguard, xp, nullptr, nullptr, nullptr, nullptr, MM, REDC, KP_SPATIAL);

    // --- GEMM 2: conv1d as gathered GEMM  M=6400 K=1280 N=256 ---
    wmma_gemm_kernel<AMODE_CONV, EPI_RELU_F16><<<dim3(REDC / 64, MM / 128), 256, 0, stream>>>(
        xp, convP, conv_b, zguard, cvo, nullptr, nullptr, nullptr, nullptr, MM, REDC, REDC * 5);

    // --- GEMM 3: drive + spikes  M=6400 K=256 N=4096 ---
    wmma_gemm_kernel<AMODE_DIRECT, EPI_DRIVE><<<dim3(WID / 64, MM / 128), 256, 0, stream>>>(
        cvo, WqF1, fc1_b, zguard, nullptr, drive_out, spikes_out, latent, gain, MM, WID, REDC);

    // --- decay-weighted reduction over T ---
    reduce_spikes_kernel<<<(BB * WID + 255) / 256, 256, 0, stream>>>(spikes_out, weighted);

    // --- logits ---
    logits_kernel<<<BB * CLS, 256, 0, stream>>>(weighted, fc2_W, fc2_b, absmax, logits);

    (void)in_sizes; (void)n_in; (void)out_size; (void)ws_size;
}
